// MultiheadAttention_8297876815996
// MI455X (gfx1250) — matrix operand order repaired and checked by reading, not compiled, not run on hardware
//
#include <hip/hip_runtime.h>
#include <hip/hip_bf16.h>

// ---------------------------------------------------------------------------
// MI455X (gfx1250) multi-head attention, bf16 WMMA pipeline, fp32 accumulate.
// B=4, S=2048, D=1024, H=16, HD=64
// Attention computed transposed (S^T = K Q^T, O^T = V^T P^T) so softmax
// reductions are in-lane + one shfl_xor(16), and P^T is built in registers.
// ---------------------------------------------------------------------------

typedef __bf16 bf16_t;
typedef __bf16 v16bf __attribute__((ext_vector_type(16)));
typedef __bf16 v2bf  __attribute__((ext_vector_type(2)));
typedef float  v8f   __attribute__((ext_vector_type(8)));

struct U4x2 { uint4 a, b; };

__device__ __forceinline__ v16bf frag_from(uint4 a, uint4 b) {
  U4x2 u; u.a = a; u.b = b;
  return __builtin_bit_cast(v16bf, u);
}

// Packs two f32 into one dword of bf16 pairs; lowers to v_cvt_pk_bf16_f32.
__device__ __forceinline__ unsigned pack_bf16x2(float a, float b) {
  v2bf v = {(bf16_t)a, (bf16_t)b};
  return __builtin_bit_cast(unsigned, v);
}

#define WMMA_BF16(A_, B_, C_) \
  __builtin_amdgcn_wmma_f32_16x16x32_bf16(false, (A_), false, (B_), (short)0, (C_), false, false)

// ---------------------------------------------------------------------------
// Kernel 0: fp32 -> bf16 conversion (grid-stride)
// ---------------------------------------------------------------------------
__global__ void cvt_f32_to_bf16(const float* __restrict__ src,
                                bf16_t* __restrict__ dst, int n) {
  int i = blockIdx.x * blockDim.x + threadIdx.x;
  int stride = gridDim.x * blockDim.x;
  for (; i < n; i += stride) dst[i] = (bf16_t)src[i];
}

// ---------------------------------------------------------------------------
// Kernel 1/3: C[M,N] = A[M,K] * Bw[N,K]^T + bias, bf16 in, fp32 accum.
// Block: 256 threads = 8 waves (2 M x 4 N). Block tile 64x256, wave tile 32x64.
// K staged through LDS in chunks of 32, incremental global pointers.
// SCALE_Q: fold the attention 1/sqrt(HD) into the q-columns (col%192 < 64).
// ---------------------------------------------------------------------------
template <bool STORE_BF16, bool SCALE_Q>
__global__ __launch_bounds__(256) void gemm_bf16_nt(
    const bf16_t* __restrict__ A,    // [M,K] row-major
    const bf16_t* __restrict__ Bw,   // [N,K] row-major
    const float*  __restrict__ bias, // [N]
    void* __restrict__ Cout,         // [M,N]
    int M, int N, int K) {
  constexpr int BM = 64, BN = 256, BK = 32, LDP = 40;  // LDP: padded row (bf16)
  __shared__ bf16_t As[BM][LDP];
  __shared__ bf16_t Bs[BN][LDP];

  const int tid  = threadIdx.x;
  const int w    = tid >> 5;
  const int lane = tid & 31;
  const int lo   = lane & 15;
  const int hi   = lane >> 4;
  const int wm   = w >> 2;   // 0..1
  const int wn   = w & 3;    // 0..3
  const int m0   = blockIdx.y * BM;
  const int n0   = blockIdx.x * BN;

  // Incremental staging pointers (advance by BK each K-step).
  const bf16_t* aSrc = A + (size_t)(m0 + (tid >> 2)) * K + (tid & 3) * 8;
  const bf16_t* bSrc = Bw + (size_t)(n0 + tid) * K;
  // Fixed LDS staging destinations.
  uint4* aDst = (uint4*)&As[tid >> 2][(tid & 3) * 8];
  uint4* bDst = (uint4*)&Bs[tid][0];
  // Fixed LDS fragment sources per wave.
  const uint4* aF0 = (const uint4*)&As[wm * 32 + lo][16 * hi];
  const uint4* aF1 = (const uint4*)&As[wm * 32 + 16 + lo][16 * hi];
  const uint4* bF[4];
#pragma unroll
  for (int ni = 0; ni < 4; ni++)
    bF[ni] = (const uint4*)&Bs[wn * 64 + ni * 16 + lo][16 * hi];

  v8f acc[2][4];
#pragma unroll
  for (int i = 0; i < 2; i++)
#pragma unroll
    for (int j = 0; j < 4; j++)
      acc[i][j] = (v8f){0.f, 0.f, 0.f, 0.f, 0.f, 0.f, 0.f, 0.f};

  for (int k0 = 0; k0 < K; k0 += BK) {
    __syncthreads();
    // Stage A tile (64 x 32): one 16B chunk per thread.
    aDst[0] = *(const uint4*)aSrc;
    // Stage B tile (256 x 32): one full row (4 x 16B) per thread.
    {
      uint4 v0 = ((const uint4*)bSrc)[0];
      uint4 v1 = ((const uint4*)bSrc)[1];
      uint4 v2 = ((const uint4*)bSrc)[2];
      uint4 v3 = ((const uint4*)bSrc)[3];
      bDst[0] = v0; bDst[1] = v1; bDst[2] = v2; bDst[3] = v3;  // stride 8 bf16
    }
    aSrc += BK;
    bSrc += BK;
    __syncthreads();

    v16bf af[2], bfr[4];
    af[0] = frag_from(aF0[0], aF0[1]);  // +8 bf16 = next uint4
    af[1] = frag_from(aF1[0], aF1[1]);
#pragma unroll
    for (int ni = 0; ni < 4; ni++)
      bfr[ni] = frag_from(bF[ni][0], *(const uint4*)((const bf16_t*)bF[ni] + 8));
#pragma unroll
    for (int mi = 0; mi < 2; mi++)
#pragma unroll
      for (int ni = 0; ni < 4; ni++)
        acc[mi][ni] = WMMA_BF16(af[mi], bfr[ni], acc[mi][ni]);
  }

  // Epilogue: C layout -> lane holds col n=lo, rows m = hi*8 + j.
#pragma unroll
  for (int mi = 0; mi < 2; mi++) {
#pragma unroll
    for (int ni = 0; ni < 4; ni++) {
      const int col = n0 + wn * 64 + ni * 16 + lo;
      const float bv = bias[col];
      float sc = 1.0f;
      if (SCALE_Q) sc = ((col % 192) < 64) ? 0.125f : 1.0f;  // 1/sqrt(64) on q cols
#pragma unroll
      for (int j = 0; j < 8; j++) {
        const int row = m0 + wm * 32 + mi * 16 + hi * 8 + j;
        const float v = (acc[mi][ni][j] + bv) * sc;
        if (STORE_BF16)
          ((bf16_t*)Cout)[(size_t)row * N + col] = (bf16_t)v;
        else
          ((float*)Cout)[(size_t)row * N + col] = v;
      }
    }
  }
}

// ---------------------------------------------------------------------------
// Kernel 2: flash attention (transposed), bf16 WMMA, online softmax.
// qkv layout: [B*S, 3072], per-row [h*192 + (q:0..63 | k:64..127 | v:128..191)]
// Block: 256 threads = 8 waves; each wave owns 16 queries; grid (S/128, B*H).
// Lane = query column; keys live in VGPR elements -> softmax reductions are
// in-lane trees plus a single shfl_xor(16). P^T built in registers.
// All global K/V pointers are loop-carried (advance by 32*ld per tile).
// ---------------------------------------------------------------------------
__global__ __launch_bounds__(256) void flash_attn_bf16(
    const bf16_t* __restrict__ qkv,   // [B*S, 3072], q-columns pre-scaled
    bf16_t* __restrict__ vals) {      // [B*S, 1024]
  constexpr int Sn = 2048, Dm = 1024, LDP = 40;
  __shared__ bf16_t Vs[64][LDP];       // V^T tile: [dim][key]

  const int tid  = threadIdx.x;
  const int w    = tid >> 5;
  const int lane = tid & 31;
  const int lo   = lane & 15;
  const int hi   = lane >> 4;
  const int b    = blockIdx.y >> 4;
  const int h    = blockIdx.y & 15;
  const int q0   = blockIdx.x * 128 + w * 16;
  const size_t ld = 3072;
  const size_t tileStep = 32 * ld;     // advance per 32-key tile

  // Q as B fragments (dims x 16 queries): lane holds query q0+lo,
  // elements = dims [kc*32 + 16*hi, +16) contiguous.
  const bf16_t* qrow = qkv + (size_t)(b * Sn + q0 + lo) * ld + h * 192;
  v16bf qf[2];
#pragma unroll
  for (int kc = 0; kc < 2; kc++)
    qf[kc] = frag_from(*(const uint4*)(qrow + kc * 32 + 16 * hi),
                       *(const uint4*)(qrow + kc * 32 + 16 * hi + 8));

  float mstate = -1e30f, lsum = 0.f;
  v8f o[4];  // O^T tiles: lane = query, elements = dims nt*16 + hi*8 + j
#pragma unroll
  for (int nt = 0; nt < 4; nt++)
    o[nt] = (v8f){0.f, 0.f, 0.f, 0.f, 0.f, 0.f, 0.f, 0.f};

  const bf16_t* kvbase = qkv + (size_t)b * Sn * ld + h * 192;
  // Loop-carried global pointers.
  const bf16_t* kr0 = kvbase + (size_t)lo * ld + 64;         // K rows, keys lo
  const bf16_t* kr1 = kvbase + (size_t)(16 + lo) * ld + 64;  // K rows, keys 16+lo
  const bf16_t* vsrc = kvbase + (size_t)w * ld + 128 + 2 * lane;  // V staging
  const bf16_t* pfp  = kvbase + (size_t)(32 + lane) * ld + 64;    // prefetch
  // Fixed LDS staging destinations: rows 2*lane, 2*lane+1, cols w + 8i.
  bf16_t* vDst0 = &Vs[2 * lane][w];
  bf16_t* vDst1 = &Vs[2 * lane + 1][w];
  // Fixed LDS fragment sources (A operand of O^T): dim row nt*16+lo.
  const uint4* vF[4];
#pragma unroll
  for (int nt = 0; nt < 4; nt++)
    vF[nt] = (const uint4*)&Vs[nt * 16 + lo][16 * hi];

  for (int kt = 0; kt < Sn; kt += 32) {
    // Prefetch next K/V tile rows (one key row per lane).
    if (kt + 32 < Sn) {
      __builtin_prefetch(pfp, 0, 1);        // K row (128B)
      __builtin_prefetch(pfp + 64, 0, 1);   // V row (128B)
    }
    __syncthreads();
    // Cooperative stage of V^T (32 keys x 64 dims) into LDS.
#pragma unroll
    for (int i = 0; i < 4; i++) {
      unsigned u = *(const unsigned*)(vsrc + (size_t)(8 * i) * ld);
      v2bf pr = __builtin_bit_cast(v2bf, u);
      vDst0[8 * i] = pr[0];
      vDst1[8 * i] = pr[1];
    }
    __syncthreads();

    // S^T = K x Q^T. c0: keys kt + hi*8 + j; c1: keys kt+16 + hi*8 + j.
    v8f c0 = (v8f){0.f, 0.f, 0.f, 0.f, 0.f, 0.f, 0.f, 0.f};
    v8f c1 = c0;
#pragma unroll
    for (int kc = 0; kc < 2; kc++) {
      v16bf kf0 = frag_from(*(const uint4*)(kr0 + kc * 32 + 16 * hi),
                            *(const uint4*)(kr0 + kc * 32 + 16 * hi + 8));
      v16bf kf1 = frag_from(*(const uint4*)(kr1 + kc * 32 + 16 * hi),
                            *(const uint4*)(kr1 + kc * 32 + 16 * hi + 8));
      c0 = WMMA_BF16(kf0, qf[kc], c0);
      c1 = WMMA_BF16(kf1, qf[kc], c1);
    }
    kr0 += tileStep; kr1 += tileStep; vsrc += tileStep; pfp += tileStep;

    // Online softmax, per-lane scalar state (lane = query).
    float tm = fmaxf(c0[0], c1[0]);
#pragma unroll
    for (int j = 1; j < 8; j++) tm = fmaxf(tm, fmaxf(c0[j], c1[j]));
    tm = fmaxf(tm, __shfl_xor(tm, 16, 32));        // join the two key halves
    const float mnew = fmaxf(mstate, tm);
    const float corr = __expf(mstate - mnew);
    mstate = mnew;

    float p0[8], p1[8];
    float rs = 0.f;
#pragma unroll
    for (int j = 0; j < 8; j++) {
      p0[j] = __expf(c0[j] - mnew);
      p1[j] = __expf(c1[j] - mnew);
      rs += p0[j] + p1[j];
    }
    rs += __shfl_xor(rs, 16, 32);
    lsum = lsum * corr + rs;
#pragma unroll
    for (int nt = 0; nt < 4; nt++)
#pragma unroll
      for (int j = 0; j < 8; j++) o[nt][j] *= corr;

    // Build P^T B fragment (32 keys x 16 queries) in registers.
    // Lane needs keys [16*hi, 16*hi+16): half is local, half from lane^16.
    unsigned pk0[4], pk1[4];
#pragma unroll
    for (int v = 0; v < 4; v++) {
      pk0[v] = pack_bf16x2(p0[2 * v], p0[2 * v + 1]);  // keys hi*8 + 2v..2v+1
      pk1[v] = pack_bf16x2(p1[2 * v], p1[2 * v + 1]);  // keys 16+hi*8 + 2v..2v+1
    }
    uint4 lowu, highu;
    {
      unsigned z[4];
#pragma unroll
      for (int v = 0; v < 4; v++) {
        const unsigned y = hi ? pk0[v] : pk1[v];       // what the partner needs
        z[v] = __shfl_xor(y, 16, 32);                  // partner's other half
      }
      lowu  = hi ? make_uint4(z[0], z[1], z[2], z[3])
                 : make_uint4(pk0[0], pk0[1], pk0[2], pk0[3]);
      highu = hi ? make_uint4(pk1[0], pk1[1], pk1[2], pk1[3])
                 : make_uint4(z[0], z[1], z[2], z[3]);
    }
    const v16bf pf = frag_from(lowu, highu);

    // O^T += V^T x P^T  (A = V^T tile from LDS, 16 dims x 32 keys).
#pragma unroll
    for (int nt = 0; nt < 4; nt++) {
      v16bf vf = frag_from(vF[nt][0],
                           *(const uint4*)((const bf16_t*)vF[nt] + 8));
      o[nt] = WMMA_BF16(vf, pf, o[nt]);
    }
  }

  // Epilogue: normalize, pack to bf16, one 16B store per (nt) tile.
  const float inv = 1.0f / lsum;
  bf16_t* orow = vals + (size_t)(b * Sn + q0 + lo) * Dm + h * 64;
#pragma unroll
  for (int nt = 0; nt < 4; nt++) {
    uint4 st;
    st.x = pack_bf16x2(o[nt][0] * inv, o[nt][1] * inv);
    st.y = pack_bf16x2(o[nt][2] * inv, o[nt][3] * inv);
    st.z = pack_bf16x2(o[nt][4] * inv, o[nt][5] * inv);
    st.w = pack_bf16x2(o[nt][6] * inv, o[nt][7] * inv);
    *(uint4*)(orow + nt * 16 + hi * 8) = st;
  }
}

// ---------------------------------------------------------------------------
// Launch
// ---------------------------------------------------------------------------
extern "C" void kernel_launch(void* const* d_in, const int* in_sizes, int n_in,
                              void* d_out, int out_size, void* d_ws, size_t ws_size,
                              hipStream_t stream) {
  constexpr int Bb = 4, Sn = 2048, Dm = 1024;
  constexpr int NR = Bb * Sn;       // 8192 rows
  constexpr int E3 = 3 * Dm;        // 3072

  const float* x      = (const float*)d_in[0];
  const float* w_qkv  = (const float*)d_in[1];
  const float* b_qkv  = (const float*)d_in[2];
  const float* w_o    = (const float*)d_in[3];
  const float* b_o    = (const float*)d_in[4];
  float* out = (float*)d_out;

  char* ws = (char*)d_ws;
  size_t off = 0;
  auto alloc = [&](size_t bytes) -> void* {
    void* p = ws + off;
    off += (bytes + 255) & ~(size_t)255;
    return p;
  };
  bf16_t* xb    = (bf16_t*)alloc((size_t)NR * Dm * 2);   // x in bf16
  bf16_t* wqkvb = (bf16_t*)alloc((size_t)E3 * Dm * 2);   // w_qkv in bf16
  bf16_t* wob   = (bf16_t*)alloc((size_t)Dm * Dm * 2);   // w_o in bf16
  bf16_t* qkvb  = (bf16_t*)alloc((size_t)NR * E3 * 2);   // qkv activations
  bf16_t* valsb = (bf16_t*)alloc((size_t)NR * Dm * 2);   // attention output

  // 0) fp32 -> bf16 conversions
  cvt_f32_to_bf16<<<4096, 256, 0, stream>>>(x, xb, NR * Dm);
  cvt_f32_to_bf16<<<4096, 256, 0, stream>>>(w_qkv, wqkvb, E3 * Dm);
  cvt_f32_to_bf16<<<2048, 256, 0, stream>>>(w_o, wob, Dm * Dm);

  // 1) QKV projection: qkv = x @ w_qkv^T + b_qkv (bf16, q-columns pre-scaled)
  gemm_bf16_nt<true, true><<<dim3(E3 / 256, NR / 64), 256, 0, stream>>>(
      xb, wqkvb, b_qkv, qkvb, NR, E3, Dm);

  // 2) Flash attention per (b,h), 128 queries per block
  flash_attn_bf16<<<dim3(Sn / 128, Bb * 16), 256, 0, stream>>>(qkvb, valsb);

  // 3) Output projection: out = vals @ w_o^T + b_o  (store fp32)
  gemm_bf16_nt<false, false><<<dim3(Dm / 256, NR / 64), 256, 0, stream>>>(
      valsb, wob, b_o, out, NR, Dm, Dm);
}
